// AttentionModel_74586402063275
// MI455X (gfx1250) — compile-verified, matrix operand order checked by reading, not run on hardware
//
#include <hip/hip_runtime.h>
#include <math.h>

#define DEV __device__ __forceinline__

typedef __attribute__((ext_vector_type(16))) __bf16 v16bf;
typedef __attribute__((ext_vector_type(8)))  __bf16 v8bf;
typedef __attribute__((ext_vector_type(8)))  float  v8f;

// ---------------------------------------------------------------------------
// CDNA5 WMMA bf16 layout helpers (wave32, V_WMMA_F32_16X16X32_BF16).
// Slot s (0..15) of a lane holds element K = k_slot(lane,s) of row (lane&15).
// Packed operand layout: [tile][kchunk][lane*16 + s]  (32B/lane contiguous).
// ---------------------------------------------------------------------------
DEV int wmma_k_slot(int lane, int s) {
  int j = s >> 1;
  return ((j & 3) << 1) + ((j >> 2) << 4) + ((lane >> 4) << 3) + (s & 1);
}

// inverse: element (row m, k-in-chunk kl) -> packed position within 512 chunk
DEV int wmma_pack_pos(int m, int kl) {
  int lane = (m & 15) | (((kl >> 3) & 1) << 4);
  int j = ((kl >> 1) & 3) | (((kl >> 4) & 1) << 2);
  int s = (j << 1) | (kl & 1);
  return lane * 16 + s;
}

DEV v8f wmma32(v16bf a, v16bf b, v8f acc) {
  return __builtin_amdgcn_wmma_f32_16x16x32_bf16(false, a, false, b,
                                                 (short)0, acc, false, false);
}

DEV float sigm(float x) { return 1.0f / (1.0f + expf(-x)); }

// ---------------------------------------------------------------------------
// Packing kernels (run every call; deterministic)
// ---------------------------------------------------------------------------
// Generic f32 matrix (N x K, row-major) -> packed bf16 tiles, dst chunk stride
// KCtot with chunk offset kcoff (lets Wi|Wh share one packed buffer).
__global__ void pack_mat_bf16(const float* __restrict__ W, __bf16* __restrict__ out,
                              int K, int KC, int KCtot, int kcoff, long total) {
  long idx = (long)blockIdx.x * 256 + threadIdx.x;
  if (idx >= total) return;
  int e = (int)(idx & 511);
  long rest = idx >> 9;
  int kc = (int)(rest % KC);
  int tile = (int)(rest / KC);
  int lane = e >> 4, s = e & 15;
  int n = tile * 16 + (lane & 15);
  int kg = kc * 32 + wmma_k_slot(lane, s);
  out[((long)tile * KCtot + kcoff + kc) * 512 + e] = (__bf16)W[(long)n * K + kg];
}

// Conv weights (Co,Ci,3,3) -> packed with K reordered to k = tap*Ci + ci.
__global__ void pack_convw_bf16(const float* __restrict__ cw, __bf16* __restrict__ out,
                                int Ci, int lc, int KC, long total) {
  long idx = (long)blockIdx.x * 256 + threadIdx.x;
  if (idx >= total) return;
  int e = (int)(idx & 511);
  long rest = idx >> 9;
  int kc = (int)(rest % KC);
  int tile = (int)(rest / KC);
  int lane = e >> 4, s = e & 15;
  int co = tile * 16 + (lane & 15);
  int kg = kc * 32 + wmma_k_slot(lane, s);
  int tap = kg >> lc;
  int ci = kg & (Ci - 1);
  out[idx] = (__bf16)cw[((long)co * Ci + ci) * 9 + tap];
}

// stage-3 activations (padded channel-last (B,6,66,512) bf16) -> encoder A
// packed per time step: out[(t*64+kc)*512+e]; feature kg -> (c=kg>>2, h=kg&3).
__global__ void pack_feats_bf16(const __bf16* __restrict__ a3, __bf16* __restrict__ out) {
  long idx = (long)blockIdx.x * 256 + threadIdx.x;
  if (idx >= 2097152L) return;
  int e = (int)(idx & 511);
  long rest = idx >> 9;
  int kc = (int)(rest & 63);
  int t = (int)(rest >> 6);
  int lane = e >> 4, s = e & 15;
  int b = lane & 15;
  int kg = kc * 32 + wmma_k_slot(lane, s);
  int c = kg >> 2, hh = kg & 3;
  out[idx] = a3[(((long)b * 6 + hh + 1) * 66 + (t + 1)) * 512 + c];
}

__global__ void fill_bf16_zero(__bf16* __restrict__ p, long n) {
  long i = (long)blockIdx.x * 256 + threadIdx.x;
  if (i < n) p[i] = (__bf16)0.0f;
}

// fused BN params: fs = g*rsqrt(v+eps); fsh = (cb-mean)*fs + beta
__global__ void bn_prep(const float* __restrict__ g, const float* __restrict__ be,
                        const float* __restrict__ mn, const float* __restrict__ vr,
                        const float* __restrict__ cb, float* __restrict__ fs,
                        float* __restrict__ fsh, int C) {
  int i = blockIdx.x * 256 + threadIdx.x;
  if (i < C) {
    float s = g[i] * rsqrtf(vr[i] + 1e-5f);
    fs[i] = s;
    fsh[i] = (cb[i] - mn[i]) * s + be[i];
  }
}

// ---------------------------------------------------------------------------
// Stage 0: Ci=1 direct conv+BN+ReLU+pool (pool folded: patch-quad sums).
// Output: padded channel-last bf16 (16, 34, 514, 64).
// ---------------------------------------------------------------------------
__global__ __launch_bounds__(256)
void conv_stage0(const float* __restrict__ mel, const float* __restrict__ w,
                 const float* __restrict__ fs, const float* __restrict__ fsh,
                 __bf16* __restrict__ out) {
  __shared__ float sw[576];
  __shared__ float sfs[64], sfsh[64];
  int tid = threadIdx.x;
  for (int i = tid; i < 576; i += 256) sw[i] = w[i];
  if (tid < 64) { sfs[tid] = fs[tid]; sfsh[tid] = fsh[tid]; }
  __syncthreads();
  long pid = (long)blockIdx.x * 256 + tid;      // 16*32*512 pooled pixels
  if (pid >= 262144L) return;
  int pw = (int)(pid & 511);
  int ph = (int)((pid >> 9) & 31);
  int b = (int)(pid >> 14);
  float patch[4][4];
#pragma unroll
  for (int yy = 0; yy < 4; ++yy)
#pragma unroll
    for (int xx = 0; xx < 4; ++xx) {
      int hh = 2 * ph - 1 + yy, ww = 2 * pw - 1 + xx;
      patch[yy][xx] = ((unsigned)hh < 64u && (unsigned)ww < 1024u)
                          ? mel[((long)b * 64 + hh) * 1024 + ww] : 0.0f;
    }
  float qsum[3][3];
#pragma unroll
  for (int dy = 0; dy < 3; ++dy)
#pragma unroll
    for (int dx = 0; dx < 3; ++dx)
      qsum[dy][dx] = patch[dy][dx] + patch[dy][dx + 1] +
                     patch[dy + 1][dx] + patch[dy + 1][dx + 1];
  long obase = (((long)b * 34 + ph + 1) * 514 + pw + 1) * 64;
  for (int cg = 0; cg < 8; ++cg) {
    v8bf ov;
#pragma unroll
    for (int r = 0; r < 8; ++r) {
      int co = cg * 8 + r;
      float acc = 0.0f;
#pragma unroll
      for (int dy = 0; dy < 3; ++dy)
#pragma unroll
        for (int dx = 0; dx < 3; ++dx)
          acc += sw[co * 9 + dy * 3 + dx] * qsum[dy][dx];
      float val = 0.25f * acc * sfs[co] + sfsh[co];
      ov[r] = (__bf16)fmaxf(val, 0.0f);
    }
    *(v8bf*)(out + obase + cg * 8) = ov;
  }
}

// ---------------------------------------------------------------------------
// Conv stages 1-3: implicit GEMM, packed bf16 weights (A, M=Co), halo-padded
// channel-last bf16 activations (B, N=pooled pixels, 4 quadrant accumulators).
// Per K-chunk: tap uniform; B = two contiguous 16B loads per lane.
// ---------------------------------------------------------------------------
__global__ __launch_bounds__(32)
void conv_pk(const __bf16* __restrict__ in, const __bf16* __restrict__ apk,
             const float* __restrict__ fs, const float* __restrict__ fsh,
             __bf16* __restrict__ out,
             int Ci, int CC, int KC, int Co, int Hp, int Wp) {
  const int lane = threadIdx.x;
  const int mtile = blockIdx.y;
  const int Hinp = Hp * 2 + 2, Winp = Wp * 2 + 2;
  const int p = blockIdx.x * 16 + (lane & 15);
  const int pw = p % Wp;
  const int t2 = p / Wp;
  const int ph = t2 % Hp;
  const int b = t2 / Hp;
  const int run0 = (lane >> 4) << 3;

  v8f acc[4];
#pragma unroll
  for (int q = 0; q < 4; ++q) acc[q] = (v8f){0.f,0.f,0.f,0.f,0.f,0.f,0.f,0.f};

  const __bf16* abase = apk + (long)mtile * KC * 512 + lane * 16;
#pragma unroll
  for (int tap = 0; tap < 9; ++tap) {
    const int dy = tap / 3, dx = tap % 3;
    const __bf16* bp[4];
#pragma unroll
    for (int q = 0; q < 4; ++q) {
      int row = 2 * ph + (q >> 1) + dy;
      int col = 2 * pw + (q & 1) + dx;
      bp[q] = in + (((long)b * Hinp + row) * Winp + col) * Ci + run0;
    }
    const __bf16* ap = abase + (long)tap * CC * 512;
    for (int cc = 0; cc < CC; ++cc) {
      v16bf av = *(const v16bf*)ap;
      ap += 512;
#pragma unroll
      for (int q = 0; q < 4; ++q) {
        const __bf16* pp = bp[q] + cc * 32;
        v8bf lo = *(const v8bf*)pp;
        v8bf hi = *(const v8bf*)(pp + 16);
        v16bf bv = __builtin_shufflevector(lo, hi, 0,1,2,3,4,5,6,7,
                                           8,9,10,11,12,13,14,15);
        acc[q] = wmma32(av, bv, acc[q]);
      }
    }
  }
  const int cobase = mtile * 16 + run0;
  v8bf ov;
#pragma unroll
  for (int r = 0; r < 8; ++r) {
    float v4 = 0.25f * (acc[0][r] + acc[1][r] + acc[2][r] + acc[3][r]);
    float val = v4 * fs[cobase + r] + fsh[cobase + r];
    ov[r] = (__bf16)fmaxf(val, 0.0f);
  }
  *(v8bf*)(out + (((long)b * (Hp + 2) + ph + 1) * (Wp + 2) + pw + 1) * Co + cobase) = ov;
}

// ---------------------------------------------------------------------------
// Generic packed GEMM: C[m,n] = sum_k Apk * Bpk + bias[n]   (f32 out)
// ---------------------------------------------------------------------------
__global__ __launch_bounds__(32)
void gemm_pk(const __bf16* __restrict__ Apk, const __bf16* __restrict__ Bpk,
             const float* __restrict__ bias, float* __restrict__ C, int N, int KC) {
  const int lane = threadIdx.x;
  const int ntile = blockIdx.x, mtile = blockIdx.y;
  const __bf16* ap = Apk + (long)mtile * KC * 512 + lane * 16;
  const __bf16* bp = Bpk + (long)ntile * KC * 512 + lane * 16;
  v8f acc = (v8f){0.f,0.f,0.f,0.f,0.f,0.f,0.f,0.f};
#pragma unroll 4
  for (int kc = 0; kc < KC; ++kc) {
    v16bf av = *(const v16bf*)ap;
    v16bf bv = *(const v16bf*)bp;
    ap += 512; bp += 512;
    acc = wmma32(av, bv, acc);
  }
  const int n = ntile * 16 + (lane & 15);
  const float bs = bias[n];
#pragma unroll
  for (int r = 0; r < 8; ++r)
    C[(long)(mtile * 16 + r + ((lane >> 4) << 3)) * N + n] = acc[r] + bs;
}

// ---------------------------------------------------------------------------
// Encoder BiLSTM: grid.x=2 (dir). 1024 thr = 32 waves; x-part A from packed
// global feats, h-part A from packed bf16 LDS (re-packed each step).
// ---------------------------------------------------------------------------
__global__ __launch_bounds__(1024)
void encoder_lstm(const __bf16* __restrict__ featsA,
                  const __bf16* __restrict__ Bpk_f, const __bf16* __restrict__ Bpk_b,
                  const float* __restrict__ bi_f, const float* __restrict__ bh_f,
                  const float* __restrict__ bi_b, const float* __restrict__ bh_b,
                  float* __restrict__ enc) {
  extern __shared__ float sm[];
  float* c = sm;                              // 4096
  float* gates = sm + 4096;                   // 16384
  __bf16* hpk = (__bf16*)(sm + 20480);        // 4096 bf16 (8 chunks)
  const int dir = blockIdx.x;
  const __bf16* Bpk = dir ? Bpk_b : Bpk_f;
  const float* bi = dir ? bi_b : bi_f;
  const float* bh = dir ? bh_b : bh_f;
  const int tid = threadIdx.x, lane = tid & 31, wave = tid >> 5;

  for (int i = tid; i < 4096; i += 1024) { c[i] = 0.0f; hpk[i] = (__bf16)0.0f; }
  __syncthreads();

  for (int step = 0; step < 64; ++step) {
    const int t = dir ? (63 - step) : step;
    for (int tile = wave; tile < 64; tile += 32) {
      const __bf16* ap = featsA + (long)t * 64 * 512 + lane * 16;
      const __bf16* bp = Bpk + (long)tile * 72 * 512 + lane * 16;
      v8f acc = (v8f){0.f,0.f,0.f,0.f,0.f,0.f,0.f,0.f};
#pragma unroll 4
      for (int kc = 0; kc < 64; ++kc) {
        v16bf av = *(const v16bf*)ap;
        v16bf bv = *(const v16bf*)bp;
        ap += 512; bp += 512;
        acc = wmma32(av, bv, acc);
      }
      const __bf16* hp = hpk + lane * 16;
#pragma unroll
      for (int kc = 0; kc < 8; ++kc) {
        v16bf av = *(const v16bf*)hp;
        v16bf bv = *(const v16bf*)bp;
        hp += 512; bp += 512;
        acc = wmma32(av, bv, acc);
      }
      const int n = tile * 16 + (lane & 15);
      const float bs = bi[n] + bh[n];
#pragma unroll
      for (int r = 0; r < 8; ++r)
        gates[(r + ((lane >> 4) << 3)) * 1024 + n] = acc[r] + bs;
    }
    __syncthreads();
    for (int idx = tid; idx < 4096; idx += 1024) {
      int b = idx >> 8, j = idx & 255;
      float ig = gates[b * 1024 + j];
      float fg = gates[b * 1024 + 256 + j];
      float gg = gates[b * 1024 + 512 + j];
      float og = gates[b * 1024 + 768 + j];
      float cc = sigm(fg) * c[idx] + sigm(ig) * tanhf(gg);
      float hh = sigm(og) * tanhf(cc);
      c[idx] = cc;
      enc[(b * 64 + t) * 512 + dir * 256 + j] = hh;
      hpk[(j >> 5) * 512 + wmma_pack_pos(b, j & 31)] = (__bf16)hh;
    }
    __syncthreads();
  }
}

// ---------------------------------------------------------------------------
// Attention decoder: one workgroup, 24 steps. All GEMM A-operands are packed
// bf16 LDS buffers; B-operands are pre-packed global weights.
// LDS: c0,c1,h1f,ctxf,gates (f32) + epk,ctxpk,h0pk,h1pk (bf16) = 319,488 B.
// ---------------------------------------------------------------------------
__global__ __launch_bounds__(1024)
void decoder_kernel(const float* __restrict__ enc, const float* __restrict__ encp,
                    const __bf16* __restrict__ WdPk, const float* __restrict__ bd,
                    const float* __restrict__ Vw, const float* __restrict__ vb,
                    const int* __restrict__ captions, const float* __restrict__ emb,
                    const __bf16* __restrict__ g0B, const float* __restrict__ bi0,
                    const float* __restrict__ bh0,
                    const __bf16* __restrict__ g1B, const float* __restrict__ bi1,
                    const float* __restrict__ bh1,
                    float* __restrict__ proj_in) {
  extern __shared__ float sm[];
  float* c0 = sm;                 // 8192
  float* c1 = sm + 8192;          // 8192
  float* h1f = sm + 16384;        // 8192
  float* ctxf = sm + 24576;       // 8192
  float* gates = sm + 32768;      // 32768
  float* qv = gates;              // alias 16x512
  float* score = gates + 8192;    // alias 16x64
  __bf16* epk = (__bf16*)(sm + 65536);  // 4096  (8 chunks)
  __bf16* ctxpk = epk + 4096;           // 8192  (16 chunks)
  __bf16* h0pk = epk + 12288;           // 8192
  __bf16* h1pk = epk + 20480;           // 8192
  const int tid = threadIdx.x, lane = tid & 31, wave = tid >> 5;

  for (int i = tid; i < 16384; i += 1024) c0[i] = 0.0f;   // c0,c1
  for (int i = tid; i < 16384; i += 1024) h0pk[i] = (__bf16)0.0f;  // h0pk,h1pk
  __syncthreads();

  for (int t = 0; t < 24; ++t) {
    // ---- q = h1 @ Wd^T + bd : 32 tiles, one per wave ----
    {
      const __bf16* ap = h1pk + lane * 16;
      const __bf16* bp = WdPk + (long)wave * 16 * 512 + lane * 16;
      v8f acc = (v8f){0.f,0.f,0.f,0.f,0.f,0.f,0.f,0.f};
#pragma unroll
      for (int kc = 0; kc < 16; ++kc) {
        v16bf av = *(const v16bf*)ap;
        v16bf bv = *(const v16bf*)bp;
        ap += 512; bp += 512;
        acc = wmma32(av, bv, acc);
      }
      const int n = wave * 16 + (lane & 15);
      const float bs = bd[n];
#pragma unroll
      for (int r = 0; r < 8; ++r)
        qv[(r + ((lane >> 4) << 3)) * 512 + n] = acc[r] + bs;
    }
    __syncthreads();
    // ---- score[b][s] = tanh(encp+q).V + vb ----
    {
      const int b = tid >> 6, s = tid & 63;
      float dot = vb[0];
      for (int j = 0; j < 512; ++j)
        dot += tanhf(encp[(b * 64 + s) * 512 + j] + qv[b * 512 + j]) * Vw[j];
      score[b * 64 + s] = dot;
    }
    __syncthreads();
    if (tid < 16) {
      const int b = tid;
      float mx = -1e30f;
      for (int s = 0; s < 64; ++s) mx = fmaxf(mx, score[b * 64 + s]);
      float sum = 0.0f;
      for (int s = 0; s < 64; ++s) {
        float e = expf(score[b * 64 + s] - mx);
        score[b * 64 + s] = e;
        sum += e;
      }
      float inv = 1.0f / sum;
      for (int s = 0; s < 64; ++s) score[b * 64 + s] *= inv;
    }
    __syncthreads();
    // ---- ctx (f32 + packed) and embedding pack ----
    for (int idx = tid; idx < 8192; idx += 1024) {
      int b = idx >> 9, j = idx & 511;
      float a = 0.0f;
      for (int s = 0; s < 64; ++s)
        a += score[b * 64 + s] * enc[(b * 64 + s) * 512 + j];
      ctxf[idx] = a;
      ctxpk[(j >> 5) * 512 + wmma_pack_pos(b, j & 31)] = (__bf16)a;
    }
    for (int idx = tid; idx < 4096; idx += 1024) {
      int b = idx >> 8, k = idx & 255;
      float v = emb[(long)captions[b * 24 + t] * 256 + k];
      epk[(k >> 5) * 512 + wmma_pack_pos(b, k & 31)] = (__bf16)v;
    }
    __syncthreads();
    // ---- LSTM0 gates: 128 tiles, K-chunks: emb(8) | ctx(16) | h0(16) ----
    for (int tile = wave; tile < 128; tile += 32) {
      const __bf16* bp = g0B + (long)tile * 40 * 512 + lane * 16;
      v8f acc = (v8f){0.f,0.f,0.f,0.f,0.f,0.f,0.f,0.f};
      const __bf16* ap = epk + lane * 16;
#pragma unroll
      for (int kc = 0; kc < 8; ++kc) {
        acc = wmma32(*(const v16bf*)ap, *(const v16bf*)bp, acc);
        ap += 512; bp += 512;
      }
      ap = ctxpk + lane * 16;
#pragma unroll
      for (int kc = 0; kc < 16; ++kc) {
        acc = wmma32(*(const v16bf*)ap, *(const v16bf*)bp, acc);
        ap += 512; bp += 512;
      }
      ap = h0pk + lane * 16;
#pragma unroll
      for (int kc = 0; kc < 16; ++kc) {
        acc = wmma32(*(const v16bf*)ap, *(const v16bf*)bp, acc);
        ap += 512; bp += 512;
      }
      const int n = tile * 16 + (lane & 15);
      const float bs = bi0[n] + bh0[n];
#pragma unroll
      for (int r = 0; r < 8; ++r)
        gates[(r + ((lane >> 4) << 3)) * 2048 + n] = acc[r] + bs;
    }
    __syncthreads();
    for (int idx = tid; idx < 8192; idx += 1024) {
      int b = idx >> 9, j = idx & 511;
      float ig = gates[b * 2048 + j];
      float fg = gates[b * 2048 + 512 + j];
      float gg = gates[b * 2048 + 1024 + j];
      float og = gates[b * 2048 + 1536 + j];
      float cc = sigm(fg) * c0[idx] + sigm(ig) * tanhf(gg);
      float hh = sigm(og) * tanhf(cc);
      c0[idx] = cc;
      h0pk[(j >> 5) * 512 + wmma_pack_pos(b, j & 31)] = (__bf16)hh;
    }
    __syncthreads();
    // ---- LSTM1 gates: K-chunks h0(16) | h1(16) ----
    for (int tile = wave; tile < 128; tile += 32) {
      const __bf16* bp = g1B + (long)tile * 32 * 512 + lane * 16;
      v8f acc = (v8f){0.f,0.f,0.f,0.f,0.f,0.f,0.f,0.f};
      const __bf16* ap = h0pk + lane * 16;
#pragma unroll
      for (int kc = 0; kc < 16; ++kc) {
        acc = wmma32(*(const v16bf*)ap, *(const v16bf*)bp, acc);
        ap += 512; bp += 512;
      }
      ap = h1pk + lane * 16;
#pragma unroll
      for (int kc = 0; kc < 16; ++kc) {
        acc = wmma32(*(const v16bf*)ap, *(const v16bf*)bp, acc);
        ap += 512; bp += 512;
      }
      const int n = tile * 16 + (lane & 15);
      const float bs = bi1[n] + bh1[n];
#pragma unroll
      for (int r = 0; r < 8; ++r)
        gates[(r + ((lane >> 4) << 3)) * 2048 + n] = acc[r] + bs;
    }
    __syncthreads();
    for (int idx = tid; idx < 8192; idx += 1024) {
      int b = idx >> 9, j = idx & 511;
      float ig = gates[b * 2048 + j];
      float fg = gates[b * 2048 + 512 + j];
      float gg = gates[b * 2048 + 1024 + j];
      float og = gates[b * 2048 + 1536 + j];
      float cc = sigm(fg) * c1[idx] + sigm(ig) * tanhf(gg);
      float hh = sigm(og) * tanhf(cc);
      c1[idx] = cc;
      h1f[idx] = hh;
      h1pk[(j >> 5) * 512 + wmma_pack_pos(b, j & 31)] = (__bf16)hh;
    }
    __syncthreads();
    for (int idx = tid; idx < 16384; idx += 1024) {
      int b = idx >> 10, j = idx & 1023;
      proj_in[(t * 16 + b) * 1024 + j] =
          (j < 512) ? h1f[b * 512 + j] : ctxf[b * 512 + (j - 512)];
    }
    __syncthreads();
  }
}

// (T*B, V) -> (B, T, V)
__global__ void rearrange_out(const float* __restrict__ tmp, float* __restrict__ out) {
  long i = (long)blockIdx.x * 256 + threadIdx.x;
  if (i >= 3840000L) return;
  int v = (int)(i % 10000);
  long r = i / 10000;
  int b = (int)(r % 16);
  int t = (int)(r / 16);
  out[((long)b * 24 + t) * 10000 + v] = tmp[i];
}

// ---------------------------------------------------------------------------
extern "C" void kernel_launch(void* const* d_in, const int* in_sizes, int n_in,
                              void* d_out, int out_size, void* d_ws, size_t ws_size,
                              hipStream_t stream) {
  (void)in_sizes; (void)n_in; (void)out_size; (void)ws_size;
  const float* mel = (const float*)d_in[0];
  const int* captions = (const int*)d_in[1];
  int pi = 2;
  const float *cw[4], *cb[4], *bg[4], *bb[4], *bm[4], *bv[4];
  for (int i = 0; i < 4; ++i) {
    cw[i] = (const float*)d_in[pi++]; cb[i] = (const float*)d_in[pi++];
    bg[i] = (const float*)d_in[pi++]; bb[i] = (const float*)d_in[pi++];
    bm[i] = (const float*)d_in[pi++]; bv[i] = (const float*)d_in[pi++];
  }
  const float* eWi_f = (const float*)d_in[pi++];
  const float* eWh_f = (const float*)d_in[pi++];
  const float* ebi_f = (const float*)d_in[pi++];
  const float* ebh_f = (const float*)d_in[pi++];
  const float* eWi_b = (const float*)d_in[pi++];
  const float* eWh_b = (const float*)d_in[pi++];
  const float* ebi_b = (const float*)d_in[pi++];
  const float* ebh_b = (const float*)d_in[pi++];
  const float* att_We = (const float*)d_in[pi++];
  const float* att_be = (const float*)d_in[pi++];
  const float* att_Wd = (const float*)d_in[pi++];
  const float* att_bd = (const float*)d_in[pi++];
  const float* att_V  = (const float*)d_in[pi++];
  const float* att_vb = (const float*)d_in[pi++];
  const float* emb    = (const float*)d_in[pi++];
  const float* Wi0 = (const float*)d_in[pi++];
  const float* Wh0 = (const float*)d_in[pi++];
  const float* bi0 = (const float*)d_in[pi++];
  const float* bh0 = (const float*)d_in[pi++];
  const float* Wi1 = (const float*)d_in[pi++];
  const float* Wh1 = (const float*)d_in[pi++];
  const float* bi1 = (const float*)d_in[pi++];
  const float* bh1 = (const float*)d_in[pi++];
  const float* outW = (const float*)d_in[pi++];
  const float* outb = (const float*)d_in[pi++];

  // ---- ws bump allocator ----
  size_t off = 0;
  auto af = [&](long n) { float* p = (float*)((char*)d_ws + off);
                          off = (off + (size_t)n * 4 + 255) & ~(size_t)255; return p; };
  auto ah = [&](long n) { __bf16* p = (__bf16*)((char*)d_ws + off);
                          off = (off + (size_t)n * 2 + 255) & ~(size_t)255; return p; };
  float* enc    = af(524288);
  float* encp   = af(524288);
  float* projin = af(393216);
  float* otmp   = af(3840000);
  float* fs0 = af(64);  float* fsh0 = af(64);
  float* fs1 = af(128); float* fsh1 = af(128);
  float* fs2 = af(256); float* fsh2 = af(256);
  float* fs3 = af(512); float* fsh3 = af(512);
  __bf16* act0 = ah(17895424L);   // (16,34,514,64)
  __bf16* act1 = ah(9510912L);    // (16,18,258,128)
  __bf16* act2 = ah(5324800L);    // (16,10,130,256)
  __bf16* act3 = ah(3244032L);    // (16,6,66,512)
  __bf16* featsA = ah(2097152L);
  __bf16* cw1pk = ah(73728L);
  __bf16* cw2pk = ah(294912L);
  __bf16* cw3pk = ah(1179648L);
  __bf16* encBf = ah(2359296L);   // [64][72][512] : Wi(0..63) Wh(64..71)
  __bf16* encBb = ah(2359296L);
  __bf16* WePk = ah(262144L);
  __bf16* WdPk = ah(262144L);
  __bf16* g0B = ah(2621440L);     // [128][40][512] : Wi0(24) Wh0(16)
  __bf16* g1B = ah(2097152L);     // [128][32][512] : Wi1(16) Wh1(16)
  __bf16* oWpk = ah(10240000L);   // [625][32][512]
  __bf16* encApk = ah(524288L);
  __bf16* projApk = ah(393216L);

  // ---- BN fusion ----
  bn_prep<<<1, 256, 0, stream>>>(bg[0], bb[0], bm[0], bv[0], cb[0], fs0, fsh0, 64);
  bn_prep<<<1, 256, 0, stream>>>(bg[1], bb[1], bm[1], bv[1], cb[1], fs1, fsh1, 128);
  bn_prep<<<1, 256, 0, stream>>>(bg[2], bb[2], bm[2], bv[2], cb[2], fs2, fsh2, 256);
  bn_prep<<<2, 256, 0, stream>>>(bg[3], bb[3], bm[3], bv[3], cb[3], fs3, fsh3, 512);

  // ---- weight packing ----
  auto P = [&](const float* W, __bf16* o, int K, int KC, int KCtot, int kcoff, long tot) {
    pack_mat_bf16<<<dim3((unsigned)((tot + 255) / 256)), 256, 0, stream>>>(
        W, o, K, KC, KCtot, kcoff, tot);
  };
  pack_convw_bf16<<<288, 256, 0, stream>>>(cw[1], cw1pk, 64, 6, 18, 73728L);
  pack_convw_bf16<<<1152, 256, 0, stream>>>(cw[2], cw2pk, 128, 7, 36, 294912L);
  pack_convw_bf16<<<4608, 256, 0, stream>>>(cw[3], cw3pk, 256, 8, 72, 1179648L);
  P(eWi_f, encBf, 2048, 64, 72, 0, 2097152L);
  P(eWh_f, encBf, 256, 8, 72, 64, 262144L);
  P(eWi_b, encBb, 2048, 64, 72, 0, 2097152L);
  P(eWh_b, encBb, 256, 8, 72, 64, 262144L);
  P(att_We, WePk, 512, 16, 16, 0, 262144L);
  P(att_Wd, WdPk, 512, 16, 16, 0, 262144L);
  P(Wi0, g0B, 768, 24, 40, 0, 1572864L);
  P(Wh0, g0B, 512, 16, 40, 24, 1048576L);
  P(Wi1, g1B, 512, 16, 32, 0, 1048576L);
  P(Wh1, g1B, 512, 16, 32, 16, 1048576L);
  P(outW, oWpk, 1024, 32, 32, 0, 10240000L);

  // ---- zero halos (whole activation buffers) ----
  fill_bf16_zero<<<69904, 256, 0, stream>>>(act0, 17895424L);
  fill_bf16_zero<<<37152, 256, 0, stream>>>(act1, 9510912L);
  fill_bf16_zero<<<20800, 256, 0, stream>>>(act2, 5324800L);
  fill_bf16_zero<<<12672, 256, 0, stream>>>(act3, 3244032L);

  // ---- CNN ----
  conv_stage0<<<1024, 256, 0, stream>>>(mel, cw[0], fs0, fsh0, act0);
  conv_pk<<<dim3(4096, 8), 32, 0, stream>>>(act0, cw1pk, fs1, fsh1, act1,
                                            64, 2, 18, 128, 16, 256);
  conv_pk<<<dim3(1024, 16), 32, 0, stream>>>(act1, cw2pk, fs2, fsh2, act2,
                                             128, 4, 36, 256, 8, 128);
  conv_pk<<<dim3(256, 32), 32, 0, stream>>>(act2, cw3pk, fs3, fsh3, act3,
                                            256, 8, 72, 512, 4, 64);

  // ---- feats -> packed encoder A ----
  pack_feats_bf16<<<8192, 256, 0, stream>>>(act3, featsA);

  // ---- BiLSTM encoder ----
  encoder_lstm<<<2, 1024, 22528 * sizeof(float), stream>>>(
      featsA, encBf, encBb, ebi_f, ebh_f, ebi_b, ebh_b, enc);

  // ---- enc_proj = enc @ We^T + be ----
  P(enc, encApk, 512, 16, 16, 0, 524288L);
  gemm_pk<<<dim3(32, 64), 32, 0, stream>>>(encApk, WePk, att_be, encp, 512, 16);

  // ---- decoder ----
  decoder_kernel<<<1, 1024, 79872 * sizeof(float), stream>>>(
      enc, encp, WdPk, att_bd, att_V, att_vb, captions, emb,
      g0B, bi0, bh0, g1B, bi1, bh1, projin);

  // ---- vocab projection ----
  P(projin, projApk, 1024, 32, 32, 0, 393216L);
  gemm_pk<<<dim3(625, 24), 32, 0, stream>>>(projApk, oWpk, outb, otmp, 10000, 32);

  rearrange_out<<<15000, 256, 0, stream>>>(otmp, (float*)d_out);
}
